// SAE_44023414784151
// MI455X (gfx1250) — matrix operand order repaired and checked by reading, not compiled, not run on hardware
//
#include <hip/hip_runtime.h>

// ---------------------------------------------------------------------------
// SAE forward: latent = relu((x-bias)@W_enc^T + b_enc); top-32 sparsify;
//              recon = latent_sparse @ (decoder / ||decoder||_row) + bias
// B=8192, D=768, H=24576, K=32 (dims derived from in_sizes at launch).
// ---------------------------------------------------------------------------

typedef __attribute__((ext_vector_type(8)))  __bf16 v8bf;
typedef __attribute__((ext_vector_type(16))) __bf16 v16bf;
typedef __attribute__((ext_vector_type(8)))  float  v8f;

#define KSEL 32

// ===========================================================================
// Kernel 1: rdnorm[h] = 1 / (||decoder[h,:]||_2 + 1e-8)   (one wave per row)
// ===========================================================================
__global__ __launch_bounds__(256) void sae_rdnorm(const float* __restrict__ dec,
                                                  float* __restrict__ rd,
                                                  int H, int D) {
  const int wave = threadIdx.x >> 5;
  const int lane = threadIdx.x & 31;
  const int h = blockIdx.x * 8 + wave;
  if (h >= H) return;
  const float* r = dec + (size_t)h * D;
  float s = 0.0f;
  for (int c = lane; c < D; c += 32) {
    float v = r[c];
    s = fmaf(v, v, s);
  }
#pragma unroll
  for (int off = 16; off > 0; off >>= 1) s += __shfl_xor(s, off, 32);
  if (lane == 0) rd[h] = 1.0f / (sqrtf(s) + 1e-8f);
}

// ===========================================================================
// Kernel 2: encoder GEMM, bf16 WMMA / f32 accumulate, double-buffered LDS.
//   Block tile: 128(M=batch) x 128(N=H), K-chunks of 32 over D.
//   8 waves: wm = wave&3 (32-row slab), wn = wave>>2 (64-col slab);
//   each wave holds 2x4 = 8 16x16 f32 accumulators.
//   Global->register loads for step i+1 issue before the 8 WMMAs of step i;
//   bf16 convert + LDS store of the other buffer after compute; one barrier
//   per K-step. LDS row stride 48 elems (96B) keeps per-lane WMMA fragments
//   as contiguous, aligned 16B/32B ds_load_b128 chunks.
// ===========================================================================
#define TM 128
#define TN 128
#define KC 32
#define LSTR 48

__global__ __launch_bounds__(256) void sae_encoder(const float* __restrict__ x,
                                                   const float* __restrict__ W,
                                                   const float* __restrict__ be,
                                                   const float* __restrict__ bias,
                                                   float* __restrict__ lat,
                                                   int D, int H) {
  __shared__ __attribute__((aligned(64))) __bf16 As[2][TM][LSTR];  // (x-bias) tiles
  __shared__ __attribute__((aligned(64))) __bf16 Bs[2][TN][LSTR];  // W_enc tiles

  const int t = threadIdx.x;
  const int h0 = blockIdx.x * TN;
  const int m0 = blockIdx.y * TM;

  const int lane = t & 31;
  const int wave = t >> 5;
  const int wm = wave & 3;   // 0..3 -> M slab of 32
  const int wn = wave >> 2;  // 0..1 -> N slab of 64
  const int lo = lane >> 4;  // lane half (selects K sub-range / rows M+8)
  const int ln = lane & 15;

  // loader mapping: each thread moves 16 consecutive floats of one row
  const int lr = t >> 1;         // 0..127
  const int lc = (t & 1) * 16;   // 0 or 16

  float ra[16], rb[16];

  auto loadG = [&](int d0) {
    const float4* gp = (const float4*)(x + (size_t)(m0 + lr) * (size_t)D + d0 + lc);
    const float4* bp = (const float4*)(bias + d0 + lc);
    const float4* wp = (const float4*)(W + (size_t)(h0 + lr) * (size_t)D + d0 + lc);
#pragma unroll
    for (int q = 0; q < 4; ++q) {
      float4 fv = gp[q];
      float4 bv = bp[q];
      float4 wvv = wp[q];
      ra[4 * q + 0] = fv.x - bv.x;
      ra[4 * q + 1] = fv.y - bv.y;
      ra[4 * q + 2] = fv.z - bv.z;
      ra[4 * q + 3] = fv.w - bv.w;
      rb[4 * q + 0] = wvv.x;
      rb[4 * q + 1] = wvv.y;
      rb[4 * q + 2] = wvv.z;
      rb[4 * q + 3] = wvv.w;
    }
  };

  auto storeL = [&](int buf) {
    v8bf alo, ahi, blo, bhi;
#pragma unroll
    for (int e = 0; e < 8; ++e) {
      alo[e] = (__bf16)ra[e];
      ahi[e] = (__bf16)ra[8 + e];
      blo[e] = (__bf16)rb[e];
      bhi[e] = (__bf16)rb[8 + e];
    }
    *(v8bf*)&As[buf][lr][lc] = alo;
    *(v8bf*)&As[buf][lr][lc + 8] = ahi;
    *(v8bf*)&Bs[buf][lr][lc] = blo;
    *(v8bf*)&Bs[buf][lr][lc + 8] = bhi;
  };

  v8f acc[2][4] = {};
  const int nIter = D / KC;

  loadG(0);
  storeL(0);
  __syncthreads();

  for (int it = 0; it < nIter; ++it) {
    // prefetch next K-chunk global->regs; latency hides under the WMMAs below
    if (it + 1 < nIter) loadG((it + 1) * KC);

    const int cur = it & 1;
    // A fragments: 16x32 bf16; lane half picks K group of 16 {16lo..16lo+15}
    v16bf av[2];
#pragma unroll
    for (int mt = 0; mt < 2; ++mt) {
      const int row = wm * 32 + mt * 16 + ln;
      v8bf a0 = *(const v8bf*)&As[cur][row][16 * lo];
      v8bf a1 = *(const v8bf*)&As[cur][row][16 * lo + 8];
      av[mt] = __builtin_shufflevector(a0, a1, 0, 1, 2, 3, 4, 5, 6, 7,
                                       8, 9, 10, 11, 12, 13, 14, 15);
    }
    // B fragments: 32x16 bf16; lane = column, half picks K group of 16
#pragma unroll
    for (int nt = 0; nt < 4; ++nt) {
      const int col = wn * 64 + nt * 16 + ln;
      v16bf bv = *(const v16bf*)&Bs[cur][col][16 * lo];
#pragma unroll
      for (int mt = 0; mt < 2; ++mt) {
        acc[mt][nt] = __builtin_amdgcn_wmma_f32_16x16x32_bf16(
            false, av[mt], false, bv, (short)0, acc[mt][nt], false, false);
      }
    }

    if (it + 1 < nIter) {
      storeL((it + 1) & 1);  // other buffer: no hazard with current readers
      __syncthreads();
    }
  }

  // ---- epilogue: +b_enc, ReLU, dense store (C layout: VGPR v -> row v+8*lo)
#pragma unroll
  for (int mt = 0; mt < 2; ++mt) {
#pragma unroll
    for (int nt = 0; nt < 4; ++nt) {
#pragma unroll
      for (int v = 0; v < 8; ++v) {
        const int m = m0 + wm * 32 + mt * 16 + v + 8 * lo;
        const int n = h0 + wn * 64 + nt * 16 + ln;
        float val = acc[mt][nt][v] + be[n];
        lat[(size_t)m * (size_t)H + n] = fmaxf(val, 0.0f);
      }
    }
  }
}

// ===========================================================================
// Kernel 3: per-row top-K (exact radix select on float bits; values >= 0
// post-ReLU so uint ordering == float ordering), deterministic compaction,
// sparsified rewrite, sparse decode from L2-resident decoder.
// One 256-thread block per batch row; the whole 96 KB row is pulled into LDS
// with CDNA5 async global->LDS b128 copies (ASYNCcnt), then scanned as f4.
// ===========================================================================
__global__ __launch_bounds__(256) void sae_topk_recon(float* __restrict__ lat,
                                                      const float* __restrict__ dec,
                                                      const float* __restrict__ bias,
                                                      const float* __restrict__ rdn,
                                                      float* __restrict__ recon,
                                                      int H, int D) {
  extern __shared__ unsigned char smem_raw[];
  float* row  = (float*)smem_raw;        // H floats, at LDS byte offset 0
  int*   hist = (int*)(row + H);         // 256
  int*   gbase = hist + 256;             // 256 (prefix of >thr counts)
  int*   ebase = gbase + 256;            // 256 (prefix of ==thr counts)
  int*   ctl  = ebase + 256;             // [0]=prefix bits, [1]=remaining, [2]=total_gt
  int*   idxs = ctl + 4;                 // KSEL
  float* vals = (float*)(idxs + KSEL);   // KSEL
  float* wv   = vals + KSEL;             // KSEL

  const int tid = threadIdx.x;
  const size_t rbase = (size_t)blockIdx.x * (size_t)H;
  const int n4 = H >> 2;

  // ---- async copy row global -> LDS (no VGPR round trip), row @ LDS off 0
  {
    const unsigned long long gb = (unsigned long long)(const void*)(lat + rbase);
    for (int j = tid; j < n4; j += 256) {
      const unsigned loff = (unsigned)j * 16u;
      const unsigned long long ga = gb + (unsigned long long)j * 16ull;
      asm volatile("global_load_async_to_lds_b128 %0, %1, off"
                   :: "v"(loff), "v"(ga)
                   : "memory");
    }
    asm volatile("s_wait_asynccnt 0x0" ::: "memory");
  }
  if (tid == 0) { ctl[0] = 0; ctl[1] = KSEL; }
  __syncthreads();

  const float4* rv = (const float4*)row;

  // ---- 4-pass radix select (MSB byte first) for K-th largest bit pattern
  for (int shift = 24; shift >= 0; shift -= 8) {
    hist[tid] = 0;
    __syncthreads();
    const unsigned pfx = (unsigned)ctl[0];
    for (int j = tid; j < n4; j += 256) {
      const float4 f = rv[j];
      const float e[4] = {f.x, f.y, f.z, f.w};
#pragma unroll
      for (int c = 0; c < 4; ++c) {
        const unsigned u = __float_as_uint(e[c]);
        const bool ok = (shift == 24) || ((u >> (shift + 8)) == pfx);
        if (ok) atomicAdd(&hist[(u >> shift) & 255], 1);
      }
    }
    __syncthreads();
    if (tid == 0) {
      int rem = ctl[1];
      int accum = 0;
      int b = 255;
      for (;; --b) {
        const int c = hist[b];
        if (accum + c >= rem) break;
        accum += c;
      }
      ctl[1] = rem - accum;                      // how many ==T to keep
      ctl[0] = (int)((pfx << 8) | (unsigned)b);  // extend prefix
    }
    __syncthreads();
  }
  const unsigned thr = (unsigned)ctl[0];  // exact bits of K-th largest value
  const int quota = ctl[1];               // tie quota at threshold

  // ---- deterministic compaction: per-thread counts + serial prefix
  int ngt = 0, neq = 0;
  for (int j = tid; j < n4; j += 256) {
    const float4 f = rv[j];
    const float e[4] = {f.x, f.y, f.z, f.w};
#pragma unroll
    for (int c = 0; c < 4; ++c) {
      const unsigned u = __float_as_uint(e[c]);
      ngt += (u > thr);
      neq += (u == thr);
    }
  }
  gbase[tid] = ngt;
  ebase[tid] = neq;
  __syncthreads();
  if (tid == 0) {
    int ag = 0, ae = 0;
    for (int k = 0; k < 256; ++k) {
      const int g = gbase[k]; gbase[k] = ag; ag += g;
      const int e = ebase[k]; ebase[k] = ae; ae += e;
    }
    ctl[2] = ag;  // total strictly-greater (== KSEL - quota)
  }
  __syncthreads();

  int gpos = gbase[tid];
  int epos = ebase[tid];
  const int cntgt = ctl[2];

  // ---- rewrite sparsified row + build active list (order: j asc, c asc)
  for (int j = tid; j < n4; j += 256) {
    const float4 f = rv[j];
    const float e[4] = {f.x, f.y, f.z, f.w};
    float o[4];
#pragma unroll
    for (int c = 0; c < 4; ++c) {
      const float v = e[c];
      const unsigned u = __float_as_uint(v);
      float outv = 0.0f;
      if (u > thr) {
        idxs[gpos] = 4 * j + c; vals[gpos] = v; ++gpos;
        outv = v;
      } else if (u == thr) {
        if (epos < quota) {
          const int p = cntgt + epos;
          idxs[p] = 4 * j + c; vals[p] = v;
          outv = v;
        }
        ++epos;
      }
      o[c] = outv;
    }
    ((float4*)(lat + rbase))[j] = make_float4(o[0], o[1], o[2], o[3]);
  }
  __syncthreads();

  // ---- scale active values by decoder row norms
  if (tid < KSEL) wv[tid] = vals[tid] * rdn[idxs[tid]];
  __syncthreads();

  // ---- sparse decode: recon[b,:] = sum_k wv[k] * decoder[idx_k,:] + bias
  const size_t ob = (size_t)blockIdx.x * (size_t)D;
  const int n4d = D >> 2;
  for (int j = tid; j < n4d; j += 256) {
    const float4 bv = ((const float4*)bias)[j];
    float a0 = bv.x, a1 = bv.y, a2 = bv.z, a3 = bv.w;
#pragma unroll 8
    for (int k = 0; k < KSEL; ++k) {
      const float w = wv[k];
      const float4 dv = ((const float4*)(dec + (size_t)idxs[k] * (size_t)D))[j];
      a0 = fmaf(w, dv.x, a0);
      a1 = fmaf(w, dv.y, a1);
      a2 = fmaf(w, dv.z, a2);
      a3 = fmaf(w, dv.w, a3);
    }
    ((float4*)(recon + ob))[j] = make_float4(a0, a1, a2, a3);
  }
}

// ===========================================================================
// Host launcher
// ===========================================================================
extern "C" void kernel_launch(void* const* d_in, const int* in_sizes, int n_in,
                              void* d_out, int out_size, void* d_ws, size_t ws_size,
                              hipStream_t stream) {
  (void)n_in; (void)out_size; (void)ws_size;
  const float* x       = (const float*)d_in[0];
  const float* W_enc   = (const float*)d_in[1];
  const float* b_enc   = (const float*)d_in[2];
  const float* bias    = (const float*)d_in[3];
  const float* decoder = (const float*)d_in[4];

  const int D = in_sizes[3];              // 768
  const int H = in_sizes[2];              // 24576
  const int B = in_sizes[0] / D;          // 8192

  float* latent = (float*)d_out;                       // [B,H]
  float* recon  = latent + (size_t)B * (size_t)H;      // [B,D]
  float* rdnorm = (float*)d_ws;                        // [H]

  // 1) decoder row norms
  sae_rdnorm<<<dim3((H + 7) / 8), dim3(256), 0, stream>>>(decoder, rdnorm, H, D);

  // 2) encoder GEMM (bf16 WMMA, f32 acc, double-buffered) -> dense relu latent
  sae_encoder<<<dim3(H / TN, B / TM), dim3(256), 0, stream>>>(x, W_enc, b_enc, bias,
                                                              latent, D, H);

  // 3) top-K sparsify + sparse decode
  const size_t smem = (size_t)H * 4 + 3 * 256 * 4 + 4 * 4 + 3 * KSEL * 4;
  (void)hipFuncSetAttribute((const void*)sae_topk_recon,
                            hipFuncAttributeMaxDynamicSharedMemorySize, (int)smem);
  sae_topk_recon<<<dim3(B), dim3(256), smem, stream>>>(latent, decoder, bias, rdnorm,
                                                       recon, H, D);
}
